// RNN_16535624090490
// MI455X (gfx1250) — matrix operand order repaired and checked by reading, not compiled, not run on hardware
//
#include <hip/hip_runtime.h>

// ---------------------------------------------------------------------------
// 2-layer tanh RNN for MI455X (gfx1250), wave32 + WMMA bf16.
//   B=64, T=512, D=256, H=1024, O=10.
// Strategy:
//   * xp = X @ Wih^T + (bih+bhh)  -> parallel bf16 WMMA GEMM over all T
//   * recurrence: persistent kernel, Whh slice resident in LDS (64KB/WGP),
//     one 16x16 WMMA tile per wave per step, global grid barrier per step.
//   * final FC is tiny (64x10) -> plain VALU kernel.
// Round-6: monitor-load arg3 is a sync-scope enum -> __MEMORY_SCOPE_DEVICE.
// ---------------------------------------------------------------------------

typedef __attribute__((ext_vector_type(16))) __bf16       v16bf;
typedef __attribute__((ext_vector_type(2)))  __bf16       v2bf;
typedef __attribute__((ext_vector_type(8)))  float        v8f;
typedef __attribute__((ext_vector_type(4)))  unsigned int uv4;
typedef __attribute__((ext_vector_type(4)))  int          iv4;
typedef __attribute__((ext_vector_type(4)))  float        fv4;

// Pointer types matching __builtin_amdgcn_global_load_async_to_lds_b128:
//   (v4i addrspace(1)* src, v4i addrspace(3)* dst, imm offset, imm cpol)
typedef __attribute__((address_space(1))) iv4* gptr4;
typedef __attribute__((address_space(3))) iv4* lptr4;
typedef __attribute__((address_space(1))) int* gip;

#ifndef __MEMORY_SCOPE_DEVICE
#define __MEMORY_SCOPE_DEVICE 1
#endif

union Frag {
  v16bf          v;
  unsigned short u[16];
  unsigned int   w[8];
  uv4            q[2];
};

__device__ __forceinline__ unsigned short f2b(float f) {   // fp32 -> bf16 (RNE)
  unsigned u = __float_as_uint(f);
  unsigned r = u + 0x7FFFu + ((u >> 16) & 1u);
  return (unsigned short)(r >> 16);
}
__device__ __forceinline__ float b2f(unsigned short b) {   // bf16 -> fp32
  return __uint_as_float(((unsigned)b) << 16);
}

// Packed fp32x2 -> bf16x2 (one v_cvt_pk_bf16_f32 when available).
__device__ __forceinline__ unsigned pk2(float lo, float hi) {
#if __has_builtin(__builtin_amdgcn_cvt_pk_bf16_f32)
  v2bf r = __builtin_amdgcn_cvt_pk_bf16_f32(lo, hi);
  return __builtin_bit_cast(unsigned, r);
#else
  return (unsigned)f2b(lo) | ((unsigned)f2b(hi) << 16);
#endif
}
__device__ __forceinline__ unsigned short f2b1(float x) {  // 1-op convert
#if __has_builtin(__builtin_amdgcn_cvt_pk_bf16_f32)
  return (unsigned short)(pk2(x, x) & 0xFFFFu);
#else
  return f2b(x);
#endif
}

// Hardware tanh (V_TANH_F32 transcendental) when the builtin exists.
__device__ __forceinline__ float fast_tanh(float x) {
#if __has_builtin(__builtin_amdgcn_tanhf)
  return __builtin_amdgcn_tanhf(x);
#elif __has_builtin(__builtin_amdgcn_tanh_f32)
  return __builtin_amdgcn_tanh_f32(x);
#else
  return tanhf(x);
#endif
}

#define HAVE_ASYNC_LDS                                              \
  (__has_builtin(__builtin_amdgcn_global_load_async_to_lds_b128) && \
   __has_builtin(__builtin_amdgcn_s_wait_asynccnt))

// Device-scope poll of the barrier generation word. Prefer the gfx1250
// memory-monitor load (wakes on cacheline change) over plain polling.
__device__ __forceinline__ unsigned bar_load(unsigned* p) {
#if __has_builtin(__builtin_amdgcn_global_load_monitor_b32)
  return (unsigned)__builtin_amdgcn_global_load_monitor_b32(
      (gip)p, 0 /*imm offset*/, __MEMORY_SCOPE_DEVICE);
#else
  return *(volatile unsigned*)p;
#endif
}
__device__ __forceinline__ void bar_sleep() {
#if __has_builtin(__builtin_amdgcn_s_monitor_sleep)
  __builtin_amdgcn_s_monitor_sleep(1);
#else
  __builtin_amdgcn_s_sleep(2);
#endif
}

// ------------------------------ grid barrier -------------------------------
__device__ __forceinline__ void grid_barrier(unsigned* cnt, unsigned* gen,
                                             unsigned nblk) {
  __threadfence();
  __syncthreads();
  if (threadIdx.x == 0) {
    unsigned g = bar_load(gen);
    unsigned arrived = atomicAdd(cnt, 1u);
    if (arrived == nblk - 1u) {
      *cnt = 0u;
      __threadfence();
      atomicAdd(gen, 1u);
    } else {
      while (bar_load(gen) == g) { bar_sleep(); }
    }
  }
  __syncthreads();
  __threadfence();
}

// ------------------------- fp32 -> bf16 convert ----------------------------
__global__ void __launch_bounds__(256)
f32_to_bf16(const float* __restrict__ in, unsigned short* __restrict__ out,
            int n) {
  int i = blockIdx.x * 256 + threadIdx.x;
  if (i < n) out[i] = f2b1(in[i]);
}

// ------------------------------- init state --------------------------------
__global__ void __launch_bounds__(256)
init_state(unsigned short* hb0, unsigned short* hb1, unsigned* cnt,
           unsigned* gen) {
  int i = blockIdx.x * 256 + threadIdx.x;   // 128*256 = 32768 uints per hbuf
  if (i < 32768) {
    ((unsigned*)hb0)[i] = 0u;
    ((unsigned*)hb1)[i] = 0u;
  }
  if (i == 0) { *cnt = 0u; *gen = 0u; }
}

// ------------------------ batched input projection -------------------------
// out[r, n] = bf16( sum_k A[arow(r), k] * W[n, k] + b0[n] + b1[n] )
// WG = 256 threads (8 waves); WG tile 128x128; wave tile 32x64 (2x4 WMMA).
template <bool AF32, bool TBT>
__global__ void __launch_bounds__(256)
gemm_bias_bf16(const void* __restrict__ Araw,
               const unsigned short* __restrict__ W,
               const float* __restrict__ b0, const float* __restrict__ b1,
               unsigned short* __restrict__ out, int M, int N, int K, int Bsz,
               int T) {
  const int lane  = threadIdx.x & 31;
  const int wid   = threadIdx.x >> 5;
  const int waveM = wid & 3, waveN = wid >> 2;
  const int l16   = lane & 15, hsel = lane >> 4;
  const int mBase = blockIdx.y * 128 + waveM * 32;
  const int nBase = blockIdx.x * 128 + waveN * 64;
  const int kSelA = hsel * 8, kSelB = hsel * 8;

  size_t arow[2];
#pragma unroll
  for (int mi = 0; mi < 2; ++mi) {
    int r = mBase + mi * 16 + l16;
    if (TBT) {  // output row r = t*B+b  reads x row b*T+t
      int t = r / Bsz;
      int bb = r - t * Bsz;
      arow[mi] = (size_t)bb * T + t;
    } else {
      arow[mi] = (size_t)r;
    }
  }

  v8f acc[2][4] = {};
  for (int k0 = 0; k0 < K; k0 += 32) {
    Frag a[2];
#pragma unroll
    for (int mi = 0; mi < 2; ++mi) {
      if (AF32) {
        const float* ap = (const float*)Araw + arow[mi] * K + k0 + kSelA;
        fv4 f0 = *(const fv4*)(ap);
        fv4 f1 = *(const fv4*)(ap + 4);
        fv4 f2 = *(const fv4*)(ap + 16);
        fv4 f3 = *(const fv4*)(ap + 20);
        a[mi].w[0] = pk2(f0[0], f0[1]);
        a[mi].w[1] = pk2(f0[2], f0[3]);
        a[mi].w[2] = pk2(f1[0], f1[1]);
        a[mi].w[3] = pk2(f1[2], f1[3]);
        a[mi].w[4] = pk2(f2[0], f2[1]);
        a[mi].w[5] = pk2(f2[2], f2[3]);
        a[mi].w[6] = pk2(f3[0], f3[1]);
        a[mi].w[7] = pk2(f3[2], f3[3]);
      } else {
        const unsigned short* ap =
            (const unsigned short*)Araw + arow[mi] * K + k0 + kSelA;
        a[mi].q[0] = *(const uv4*)(ap);
        a[mi].q[1] = *(const uv4*)(ap + 16);
      }
    }
#pragma unroll
    for (int ni = 0; ni < 4; ++ni) {
      const unsigned short* bpp =
          W + (size_t)(nBase + ni * 16 + l16) * K + k0 + kSelB;
      Frag b;
      b.q[0] = *(const uv4*)(bpp);
      b.q[1] = *(const uv4*)(bpp + 16);
#pragma unroll
      for (int mi = 0; mi < 2; ++mi)
        acc[mi][ni] = __builtin_amdgcn_wmma_f32_16x16x32_bf16(
            false, a[mi].v, false, b.v, (short)0, acc[mi][ni], false, false);
    }
  }

#pragma unroll
  for (int ni = 0; ni < 4; ++ni) {
    int col = nBase + ni * 16 + l16;
    float bias = b0[col] + b1[col];
#pragma unroll
    for (int mi = 0; mi < 2; ++mi) {
#pragma unroll
      for (int i = 0; i < 8; ++i) {
        int row = mBase + mi * 16 + i + hsel * 8;
        out[(size_t)row * N + col] = f2b1(acc[mi][ni][i] + bias);
      }
    }
  }
}

// ---------------------------- recurrence (persistent) ----------------------
// h_t = tanh(xp_t + h_{t-1} @ Whh^T), h: 64x1024.
// 32 WGs x 8 waves = 256 waves; wave (mTile 0..3, nTile 0..63) owns one
// 16x16 tile. Each WG keeps its 32 Whh rows (64KB bf16) in LDS for all T.
// xp_t is loaded ahead and used as the WMMA C-input, so the serial tail per
// step is just tanh -> cvt -> store -> barrier.
template <bool STORE_SEQ>
__global__ void __launch_bounds__(256)
rnn_recurrence(const unsigned short* __restrict__ whh,   // (H,H) bf16 row-major
               const unsigned short* __restrict__ xp,    // (T*B, H) bf16
               unsigned short* __restrict__ hb0, unsigned short* __restrict__ hb1,
               unsigned short* __restrict__ seq,         // (T*B, H)
               unsigned* cnt, unsigned* gen, int T) {
  __shared__ alignas(32) unsigned short lds[32 * 1024];  // 64 KB of 320 KB/WGP
  {
    const unsigned short* src = whh + (size_t)blockIdx.x * 32u * 1024u;
#if HAVE_ASYNC_LDS
    for (int i = threadIdx.x; i < 4096; i += 256) {
      __builtin_amdgcn_global_load_async_to_lds_b128(
          (gptr4)(src + (size_t)i * 8), (lptr4)(lds + (size_t)i * 8), 0, 0);
    }
    __builtin_amdgcn_s_wait_asynccnt(0);
#else
    const uv4* s4 = (const uv4*)src;
    uv4* dst = (uv4*)lds;
    for (int i = threadIdx.x; i < 4096; i += 256) dst[i] = s4[i];
#endif
  }
  __syncthreads();

  const int lane  = threadIdx.x & 31;
  const int wid   = threadIdx.x >> 5;
  const int mTile = wid & 3;
  const int nLoc  = wid >> 2;
  const int l16   = lane & 15;
  const int hsel  = lane >> 4;
  const int rowA  = mTile * 16 + l16;
  const int kSelA = hsel * 8;
  const int colL  = nLoc * 16 + l16;            // column within LDS slice
  const int colG  = blockIdx.x * 32 + colL;     // global output column
  const int kSelB = hsel * 8;

  for (int t = 0; t < T; ++t) {
    const unsigned short* hp = (t & 1) ? hb1 : hb0;
    unsigned short* hn = (t & 1) ? hb0 : hb1;
    const size_t xrow = (size_t)t * 64u;

    // xp addend for this tile doubles as the WMMA C-input; loads overlap the
    // first h-state fragment loads (same L2 latency, issued in parallel).
    v8f acc;
#pragma unroll
    for (int i = 0; i < 8; ++i) {
      int row = mTile * 16 + i + hsel * 8;
      acc[i] = b2f(xp[(xrow + row) * 1024 + colG]);
    }

    const unsigned short* ap = hp + rowA * 1024 + kSelA;
    const unsigned short* bp = lds + colL * 1024 + kSelB;
#pragma unroll 8
    for (int k0 = 0; k0 < 1024; k0 += 32) {
      Frag a, b;
      a.q[0] = *(const uv4*)(ap + k0);
      a.q[1] = *(const uv4*)(ap + k0 + 16);
      b.q[0] = *(const uv4*)(bp + k0);
      b.q[1] = *(const uv4*)(bp + k0 + 16);
      acc = __builtin_amdgcn_wmma_f32_16x16x32_bf16(false, a.v, false, b.v,
                                                    (short)0, acc, false, false);
    }

#pragma unroll
    for (int i = 0; i < 8; ++i) {
      int row = mTile * 16 + i + hsel * 8;
      unsigned short hbits = f2b1(fast_tanh(acc[i]));
      hn[row * 1024 + colG] = hbits;
      if (STORE_SEQ) seq[(xrow + row) * 1024 + colG] = hbits;
    }
    grid_barrier(cnt, gen, gridDim.x);
  }
}

// --------------------------------- final FC --------------------------------
__global__ void __launch_bounds__(256)
fc_kernel(const unsigned short* __restrict__ hfin, const float* __restrict__ Wfc,
          const float* __restrict__ bfc, float* __restrict__ out) {
  int idx = blockIdx.x * 256 + threadIdx.x;
  if (idx >= 64 * 10) return;
  int b = idx / 10, o = idx - b * 10;
  float s = 0.f;
  for (int k = 0; k < 1024; ++k)
    s += b2f(hfin[b * 1024 + k]) * Wfc[o * 1024 + k];
  out[idx] = s + bfc[o];
}

// ------------------------------- host launch -------------------------------
extern "C" void kernel_launch(void* const* d_in, const int* in_sizes, int n_in,
                              void* d_out, int out_size, void* d_ws,
                              size_t ws_size, hipStream_t stream) {
  (void)in_sizes; (void)n_in; (void)out_size; (void)ws_size;
  const float* x    = (const float*)d_in[0];
  const float* Wih0 = (const float*)d_in[1];
  const float* Whh0 = (const float*)d_in[2];
  const float* bih0 = (const float*)d_in[3];
  const float* bhh0 = (const float*)d_in[4];
  const float* Wih1 = (const float*)d_in[5];
  const float* Whh1 = (const float*)d_in[6];
  const float* bih1 = (const float*)d_in[7];
  const float* bhh1 = (const float*)d_in[8];
  const float* Wfc  = (const float*)d_in[9];
  const float* bfc  = (const float*)d_in[10];
  float* out = (float*)d_out;

  char* ws = (char*)d_ws;
  const size_t XP  = 0;                                   // 64 MB  (T*B,H) bf16
  const size_t SEQ = XP  + (size_t)32768 * 1024 * 2;      // 64 MB  layer-1 h
  const size_t WI0 = SEQ + (size_t)32768 * 1024 * 2;      // bf16 weights
  const size_t WH0 = WI0 + (size_t)1024 * 256 * 2;
  const size_t WI1 = WH0 + (size_t)1024 * 1024 * 2;
  const size_t WH1 = WI1 + (size_t)1024 * 1024 * 2;
  const size_t HB0 = WH1 + (size_t)1024 * 1024 * 2;       // h double buffer
  const size_t HB1 = HB0 + (size_t)64 * 1024 * 2;
  const size_t BAR = HB1 + (size_t)64 * 1024 * 2;

  unsigned short* xp    = (unsigned short*)(ws + XP);
  unsigned short* seq   = (unsigned short*)(ws + SEQ);
  unsigned short* wih0b = (unsigned short*)(ws + WI0);
  unsigned short* whh0b = (unsigned short*)(ws + WH0);
  unsigned short* wih1b = (unsigned short*)(ws + WI1);
  unsigned short* whh1b = (unsigned short*)(ws + WH1);
  unsigned short* hb0   = (unsigned short*)(ws + HB0);
  unsigned short* hb1   = (unsigned short*)(ws + HB1);
  unsigned* cnt = (unsigned*)(ws + BAR);
  unsigned* gen = cnt + 64;   // keep counters on separate cachelines

  // Pre-convert weights to bf16 once per call.
  f32_to_bf16<<<dim3(1024), 256, 0, stream>>>(Wih0, wih0b, 1024 * 256);
  f32_to_bf16<<<dim3(4096), 256, 0, stream>>>(Whh0, whh0b, 1024 * 1024);
  f32_to_bf16<<<dim3(4096), 256, 0, stream>>>(Wih1, wih1b, 1024 * 1024);
  f32_to_bf16<<<dim3(4096), 256, 0, stream>>>(Whh1, whh1b, 1024 * 1024);

  // Layer 0
  gemm_bias_bf16<true, true><<<dim3(8, 256), 256, 0, stream>>>(
      x, wih0b, bih0, bhh0, xp, 32768, 1024, 256, 64, 512);
  init_state<<<dim3(128), 256, 0, stream>>>(hb0, hb1, cnt, gen);
  rnn_recurrence<true><<<dim3(32), 256, 0, stream>>>(whh0b, xp, hb0, hb1, seq,
                                                     cnt, gen, 512);

  // Layer 1
  gemm_bias_bf16<false, false><<<dim3(8, 256), 256, 0, stream>>>(
      seq, wih1b, bih1, bhh1, xp, 32768, 1024, 1024, 64, 512);
  init_state<<<dim3(128), 256, 0, stream>>>(hb0, hb1, cnt, gen);
  rnn_recurrence<false><<<dim3(32), 256, 0, stream>>>(whh1b, xp, hb0, hb1, seq,
                                                      cnt, gen, 512);

  // Final FC: T=512 (even) -> last hidden state ends in hb0.
  fc_kernel<<<dim3(3), 256, 0, stream>>>(hb0, Wfc, bfc, out);
}
